// mtrnn_75479755260395
// MI455X (gfx1250) — matrix operand order repaired and checked by reading, not compiled, not run on hardware
//
#include <hip/hip_runtime.h>

// ---------------------------------------------------------------------------
// MTRNN persistent kernel for gfx1250 (MI455X), round 2.
//  - 64 workgroups x 128 threads (4 wave32 waves); WG b owns batch rows
//    [16b, 16b+16). States (hf, cf, hs, cs) live in static LDS as f16.
//  - ALL weight fragments + bias tables (203,648 B) are staged into dynamic
//    LDS once at kernel start (WGP has 320 KB), then every one of the 512
//    steps feeds v_wmma_f32_16x16x32_f16 B-operands from LDS instead of L2.
//  - Fragment intra-layout is bank-conflict-minimal: two 512B halves, lane
//    stride 16B per ds_load_b128.
// Padded dims: K_in 28->32, N_fh 100->112 (7 tiles), K_fh 100->128 (4 chunks),
//              N_sh/K_sh 30->32, N_cls 10->16.
// ---------------------------------------------------------------------------

typedef _Float16 v16h __attribute__((ext_vector_type(16)));
typedef _Float16 v8h  __attribute__((ext_vector_type(8)));
typedef float    v8f  __attribute__((ext_vector_type(8)));

#define T_STEPS 512
#define BATCH   1024
#define N_IN    28

// fragment table: 196 fragments of 512 f16 (1KB each)
// [0,140)   fh: ((gi*7+nt)*5 + j), j=0: x2 (K=32), j=1..4: h2 chunk j-1
// [140,180) sh: 140+((gi*2+nt)*5 + j), j=0..3: x2(hf) chunk j, j=4: h2(cs)
// [180,187) hs2hf: nt
// [187,195) hf2hs: nt*4 + kc
// [195]     fc
// Intra-fragment layout (f16 index within fragment):
//   (e<8 ? 0 : 256) + lane*8 + (e&7)   -- two 512B halves, lane stride 16B
#define NFRAG 196
#define WFRAG_F16   (NFRAG * 512)              // 100352 f16 = 200704 B
#define BIAS_F32    736                        // fh[4][112] sh[4][32] p3[112] p4[32] fc[16]
#define LDSW_BYTES  (WFRAG_F16 * 2 + BIAS_F32 * 4)  // 203648
#define WS_BIAS_F32_OFF (WFRAG_F16 / 2)        // f32 index of bias region in ws

struct WPtrs {
  const float* fh_x2w[4]; const float* fh_x2b[4];
  const float* fh_h2w[4]; const float* fh_h2b[4];
  const float* sh_x2w[4]; const float* sh_x2b[4];
  const float* sh_h2w[4]; const float* sh_h2b[4];
  const float* hs2hf_w;   const float* hs2hf_b;
  const float* hf2hs_w;   const float* hf2hs_b;
  const float* fc_w;      const float* fc_b;
};

// ---------------------------------------------------------------- helpers ---
__device__ __forceinline__ v8f wmma32(v16h a, v16h b, v8f c) {
  return __builtin_amdgcn_wmma_f32_16x16x32_f16(false, a, false, b, (short)0, c,
                                                false, false);
}

__device__ __forceinline__ v8f splat8(float x) {
  v8f r;
#pragma unroll
  for (int i = 0; i < 8; ++i) r[i] = x;
  return r;
}

// A-fragment (16x32 f16) from LDS row-major [16][stride], K-chunk kc.
// Lane l: m = l&15, kb = (l>>4)*8; elems 0..7 -> k = kb+0..7,
// elems 8..15 -> k = 16+kb+0..7  (ISA 7.12.2, 16-bit A layout).
__device__ __forceinline__ v16h ldsA(const _Float16* base, int stride, int kc,
                                     int m, int kb) {
  const _Float16* p = base + m * stride + kc * 32 + kb;
  v8h lo = *(const v8h*)(p);
  v8h hi = *(const v8h*)(p + 16);
  v16h a;
#pragma unroll
  for (int i = 0; i < 8; ++i) { a[i] = lo[i]; a[i + 8] = hi[i]; }
  return a;
}

// B-fragment from LDS weight cache: two ds_load_b128, lane stride 16B.
__device__ __forceinline__ v16h bfragL(const _Float16* __restrict__ w, int frag,
                                       int lane) {
  const _Float16* p = w + frag * 512 + lane * 8;
  v8h lo = *(const v8h*)(p);
  v8h hi = *(const v8h*)(p + 256);
  v16h b;
#pragma unroll
  for (int i = 0; i < 8; ++i) { b[i] = lo[i]; b[i + 8] = hi[i]; }
  return b;
}

__device__ __forceinline__ float fsig(float x) {
  return 1.0f / (1.0f + __expf(-x));
}
__device__ __forceinline__ float ftanh(float x) {
  float e = __expf(2.0f * x);
  return 1.0f - 2.0f / (e + 1.0f);  // robust at +/-inf
}

// ------------------------------------------------------------ pack kernels --
__global__ void mtrnn_pack_weights(WPtrs P, _Float16* __restrict__ dst) {
  int gid = blockIdx.x * 256 + threadIdx.x;
  if (gid >= WFRAG_F16) return;
  int frag = gid >> 9, r = gid & 511;
  // half-split intra-layout decode
  int lane = (r >> 3) & 31;
  int e    = (r & 7) | ((r >> 8) << 3);
  int nl = lane & 15;
  int k  = ((lane >> 4) << 3) + (e & 7) + ((e >> 3) << 4);  // 0..31 within K-chunk (B layout)
  float v = 0.0f;
  if (frag < 140) {
    int q = frag / 5, j = frag - q * 5, gi = q / 7, nt = q - gi * 7;
    int n = nt * 16 + nl;
    if (n < 100) {
      if (j == 0) { if (k < 28) v = P.fh_x2w[gi][n * 28 + k]; }
      else { int kg = (j - 1) * 32 + k; if (kg < 100) v = P.fh_h2w[gi][n * 100 + kg]; }
    }
  } else if (frag < 180) {
    int f2 = frag - 140; int q = f2 / 5, j = f2 - q * 5, gi = q / 2, nt = q & 1;
    int n = nt * 16 + nl;
    if (n < 30) {
      if (j < 4) { int kg = j * 32 + k; if (kg < 100) v = P.sh_x2w[gi][n * 100 + kg]; }
      else { if (k < 30) v = P.sh_h2w[gi][n * 30 + k]; }
    }
  } else if (frag < 187) {
    int nt = frag - 180; int n = nt * 16 + nl;
    if (n < 100 && k < 30) v = P.hs2hf_w[n * 30 + k];
  } else if (frag < 195) {
    int idx = frag - 187; int nt = idx >> 2, kc = idx & 3;
    int n = nt * 16 + nl; int kg = kc * 32 + k;
    if (n < 30 && kg < 100) v = P.hf2hs_w[n * 100 + kg];
  } else {
    int n = nl;
    if (n < 10 && k < 30) v = P.fc_w[n * 30 + k];
  }
  dst[gid] = (_Float16)v;
}

// bias tables (f32): fh[4][112], sh[4][32], hs2hf[112], hf2hs[32], fc[16]
__global__ void mtrnn_pack_bias(WPtrs P, float* __restrict__ dst) {
  int i = blockIdx.x * 256 + threadIdx.x;
  if (i >= BIAS_F32) return;
  float v = 0.0f;
  if (i < 448)      { int gi = i / 112, n = i % 112; if (n < 100) v = P.fh_x2b[gi][n] + P.fh_h2b[gi][n]; }
  else if (i < 576) { int j = i - 448; int gi = j / 32, n = j % 32; if (n < 30) v = P.sh_x2b[gi][n] + P.sh_h2b[gi][n]; }
  else if (i < 688) { int n = i - 576; if (n < 100) v = P.hs2hf_b[n]; }
  else if (i < 720) { int n = i - 688; if (n < 30)  v = P.hf2hs_b[n]; }
  else              { int n = i - 720; if (n < 10)  v = P.fc_b[n]; }
  dst[i] = v;
}

// -------------------------------------------------------------- main kernel -
__global__ __launch_bounds__(128, 1)
void mtrnn_persist(const float* __restrict__ data,
                   const _Float16* __restrict__ wsrc,  // ws: frags then biases
                   float* __restrict__ out) {
  __shared__ __align__(16) _Float16 s_x [16 * 32];
  __shared__ __align__(16) _Float16 s_hf[16 * 128];
  __shared__ __align__(16) _Float16 s_cf[16 * 128];
  __shared__ __align__(16) _Float16 s_hs [16 * 32];
  __shared__ __align__(16) _Float16 s_hs1[16 * 32];
  __shared__ __align__(16) _Float16 s_cs [16 * 32];
  extern __shared__ __align__(16) char s_dynraw[];   // 203,648 B weight cache
  _Float16* s_w = (_Float16*)s_dynraw;

  const int tid  = threadIdx.x;
  const int wave = tid >> 5;
  const int lane = tid & 31;
  const int ncol = lane & 15;         // N within a 16-wide tile (C/D layout)
  const int mrow = (lane >> 4) * 8;   // C/D row base
  const int kb   = (lane >> 4) * 8;   // A-fragment K base
  const int am   = lane & 15;         // A-fragment row
  const int m0   = blockIdx.x * 16;   // batch row base for this WG

  // ---- one-time: stage all weight fragments + biases into LDS (b128 copies)
  {
    const v8h* __restrict__ src = (const v8h*)wsrc;
    v8h* dst = (v8h*)s_w;
    for (int i = tid; i < LDSW_BYTES / 16; i += 128) dst[i] = src[i];
  }
  // zero states + padding once
  for (int i = tid; i < 16 * 128; i += 128) { s_hf[i] = (_Float16)0.f; s_cf[i] = (_Float16)0.f; }
  for (int i = tid; i < 16 * 32; i += 128) {
    s_hs[i] = (_Float16)0.f; s_hs1[i] = (_Float16)0.f;
    s_cs[i] = (_Float16)0.f; s_x[i]  = (_Float16)0.f;
  }
  __syncthreads();

  const float* bias_fh = (const float*)(s_w + WFRAG_F16);  // [4][112]
  const float* bias_sh = bias_fh + 448;  // [4][32]
  const float* bias_p3 = bias_fh + 576;  // [112]
  const float* bias_p4 = bias_fh + 688;  // [32]
  const float* bias_fc = bias_fh + 720;  // [16]
  const float c23 = 2.0f / 3.0f, c13 = 1.0f / 3.0f;

  for (int t = 0; t < T_STEPS; ++t) {
    // ---- phase 0: stage x_t tile into LDS (f16), prefetch next slab ----
    const float* xp = data + ((size_t)t * BATCH + m0) * N_IN;
    for (int i = tid; i < 16 * N_IN; i += 128) {
      int r = i / N_IN, c = i - r * N_IN;
      s_x[r * 32 + c] = (_Float16)xp[r * N_IN + c];
    }
    if (tid < 16 && t + 1 < T_STEPS)
      __builtin_prefetch(data + ((size_t)(t + 1) * BATCH + m0 + tid) * N_IN, 0, 3);
    __syncthreads();  // B1: x ready

    // ---- phase 1: fast cell. A inputs: x (K=32), cf_prev (K=128) ----
    v16h a_x = ldsA(s_x, 32, 0, am, kb);
    v16h a_c[4];
#pragma unroll
    for (int kc = 0; kc < 4; ++kc) a_c[kc] = ldsA(s_cf, 128, kc, am, kb);
    __syncthreads();  // B2: A-frags captured before states are overwritten

#pragma unroll
    for (int rep = 0; rep < 2; ++rep) {
      const int nt = wave + rep * 4;           // tiles 0..6 over 4 waves
      if (nt < 7) {
        const int n = nt * 16 + ncol;
        v8f acc[4];
#pragma unroll
        for (int gi = 0; gi < 4; ++gi) acc[gi] = splat8(bias_fh[gi * 112 + n]);
#pragma unroll
        for (int gi = 0; gi < 4; ++gi) {
          const int fb = (gi * 7 + nt) * 5;
          acc[gi] = wmma32(a_x,    bfragL(s_w, fb + 0, lane), acc[gi]);
          acc[gi] = wmma32(a_c[0], bfragL(s_w, fb + 1, lane), acc[gi]);
          acc[gi] = wmma32(a_c[1], bfragL(s_w, fb + 2, lane), acc[gi]);
          acc[gi] = wmma32(a_c[2], bfragL(s_w, fb + 3, lane), acc[gi]);
          acc[gi] = wmma32(a_c[3], bfragL(s_w, fb + 4, lane), acc[gi]);
        }
        // cf_new = f*hf_prev + i*g ; hf_cell = o*tanh(cf_new)
#pragma unroll
        for (int v = 0; v < 8; ++v) {
          const int idx = (mrow + v) * 128 + n;
          float fg = fsig(acc[0][v]), ig = fsig(acc[1][v]);
          float gg = ftanh(acc[2][v]), og = fsig(acc[3][v]);
          float cn = fg * (float)s_hf[idx] + ig * gg;
          s_cf[idx] = (_Float16)cn;
          s_hf[idx] = (_Float16)(og * ftanh(cn));
        }
      }
    }
    __syncthreads();  // B3: hf_cell / cf_new visible

    // ---- phase 2: slow cell (waves 0,1). A: hf_new (K=128), cs_prev (K=32) --
    v16h a_h[4], a_cs2;
    if (wave < 2) {
#pragma unroll
      for (int kc = 0; kc < 4; ++kc) a_h[kc] = ldsA(s_hf, 128, kc, am, kb);
      a_cs2 = ldsA(s_cs, 32, 0, am, kb);
    }
    __syncthreads();  // B4: cs_prev captured before overwrite
    if (wave < 2) {
      const int nt = wave;
      const int n = nt * 16 + ncol;
      v8f acc[4];
#pragma unroll
      for (int gi = 0; gi < 4; ++gi) acc[gi] = splat8(bias_sh[gi * 32 + n]);
#pragma unroll
      for (int gi = 0; gi < 4; ++gi) {
        const int fb = 140 + (gi * 2 + nt) * 5;
        acc[gi] = wmma32(a_h[0], bfragL(s_w, fb + 0, lane), acc[gi]);
        acc[gi] = wmma32(a_h[1], bfragL(s_w, fb + 1, lane), acc[gi]);
        acc[gi] = wmma32(a_h[2], bfragL(s_w, fb + 2, lane), acc[gi]);
        acc[gi] = wmma32(a_h[3], bfragL(s_w, fb + 3, lane), acc[gi]);
        acc[gi] = wmma32(a_cs2,  bfragL(s_w, fb + 4, lane), acc[gi]);
      }
#pragma unroll
      for (int v = 0; v < 8; ++v) {
        const int idx = (mrow + v) * 32 + n;
        float fg = fsig(acc[0][v]), ig = fsig(acc[1][v]);
        float gg = ftanh(acc[2][v]), og = fsig(acc[3][v]);
        float cn = fg * (float)s_hs[idx] + ig * gg;
        s_cs[idx]  = (_Float16)cn;
        s_hs1[idx] = (_Float16)(og * ftanh(cn));
      }
    }
    __syncthreads();  // B5: hs1 / cs_new visible

    // ---- phase 3: hf blend. hf = (2/3)hf + tanh(hf + hs1@Whs2hf + b)/3 ----
    v16h a_s1 = ldsA(s_hs1, 32, 0, am, kb);
#pragma unroll
    for (int rep = 0; rep < 2; ++rep) {
      const int nt = wave + rep * 4;
      if (nt < 7) {
        const int n = nt * 16 + ncol;
        v8f acc = splat8(bias_p3[n]);
        acc = wmma32(a_s1, bfragL(s_w, 180 + nt, lane), acc);
#pragma unroll
        for (int v = 0; v < 8; ++v) {
          const int idx = (mrow + v) * 128 + n;
          float h1 = (float)s_hf[idx];
          s_hf[idx] = (_Float16)(c23 * h1 + c13 * ftanh(h1 + acc[v]));
        }
      }
    }
    __syncthreads();  // B6: hf_final visible

    // ---- phase 4: hs blend (waves 0,1). hs = .9*hs1 + .1*tanh(hs1+hf@W+b) --
    v16h a_f[4];
    if (wave < 2) {
#pragma unroll
      for (int kc = 0; kc < 4; ++kc) a_f[kc] = ldsA(s_hf, 128, kc, am, kb);
      const int nt = wave;
      const int n = nt * 16 + ncol;
      v8f acc = splat8(bias_p4[n]);
#pragma unroll
      for (int kc = 0; kc < 4; ++kc)
        acc = wmma32(a_f[kc], bfragL(s_w, 187 + nt * 4 + kc, lane), acc);
#pragma unroll
      for (int v = 0; v < 8; ++v) {
        const int idx = (mrow + v) * 32 + n;
        float h1 = (float)s_hs1[idx];
        s_hs[idx] = (_Float16)(0.9f * h1 + 0.1f * ftanh(h1 + acc[v]));
      }
    }
    __syncthreads();  // B7: hs_final visible

    // ---- phase 5: classifier head (wave 3), write logits to global ----
    if (wave == 3) {
      v16h a_hsf = ldsA(s_hs, 32, 0, am, kb);
      v8f acc = splat8(bias_fc[ncol]);
      acc = wmma32(a_hsf, bfragL(s_w, 195, lane), acc);
      if (ncol < 10) {
#pragma unroll
        for (int v = 0; v < 8; ++v) {
          const int m = mrow + v;
          out[((size_t)t * BATCH + m0 + m) * 10 + ncol] = acc[v];
        }
      }
    }
    // loop-top B1 of next iter isolates phase-5 reads from future writes
  }
}

// ------------------------------------------------------------- launch glue --
extern "C" void kernel_launch(void* const* d_in, const int* in_sizes, int n_in,
                              void* d_out, int out_size, void* d_ws,
                              size_t ws_size, hipStream_t stream) {
  const float* const* in = (const float* const*)d_in;
  WPtrs P;
  const float* data = in[0];  // data is index 0 in both flattening orders

  if (n_in > 1 && in_sizes[1] == 10) {
    // jax tree_flatten (alphabetical) order:
    // 0:data 1:fc_b 2:fc_w 3..18:fh{h2f_b,h2f_w,h2g_b,h2g_w,h2i_b,h2i_w,
    //   h2o_b,h2o_w,x2f_b,x2f_w,x2g_b,x2g_w,x2i_b,x2i_w,x2o_b,x2o_w}
    // 19:hf2hs_b 20:hf2hs_w 21:hs2hf_b 22:hs2hf_w 23..38:sh{same pattern}
    const int perm[4] = {0, 2, 1, 3};  // gate (f,i,g,o) -> alpha pos (f,g,i,o)
    P.fc_b = in[1]; P.fc_w = in[2];
    for (int gi = 0; gi < 4; ++gi) {
      int p = perm[gi];
      P.fh_h2b[gi] = in[3 + p * 2];      P.fh_h2w[gi] = in[3 + p * 2 + 1];
      P.fh_x2b[gi] = in[3 + 8 + p * 2];  P.fh_x2w[gi] = in[3 + 8 + p * 2 + 1];
      P.sh_h2b[gi] = in[23 + p * 2];     P.sh_h2w[gi] = in[23 + p * 2 + 1];
      P.sh_x2b[gi] = in[23 + 8 + p * 2]; P.sh_x2w[gi] = in[23 + 8 + p * 2 + 1];
    }
    P.hf2hs_b = in[19]; P.hf2hs_w = in[20];
    P.hs2hf_b = in[21]; P.hs2hf_w = in[22];
  } else {
    // dict-insertion order: data, fh{(x2f,h2f,x2i,h2i,x2g,h2g,x2o,h2o) w,b},
    // sh{same}, hs2hf_w/b, hf2hs_w/b, fc_w/b
    for (int gi = 0; gi < 4; ++gi) {
      int bf = 1 + gi * 4, bs = 17 + gi * 4;
      P.fh_x2w[gi] = in[bf];     P.fh_x2b[gi] = in[bf + 1];
      P.fh_h2w[gi] = in[bf + 2]; P.fh_h2b[gi] = in[bf + 3];
      P.sh_x2w[gi] = in[bs];     P.sh_x2b[gi] = in[bs + 1];
      P.sh_h2w[gi] = in[bs + 2]; P.sh_h2b[gi] = in[bs + 3];
    }
    P.hs2hf_w = in[33]; P.hs2hf_b = in[34];
    P.hf2hs_w = in[35]; P.hf2hs_b = in[36];
    P.fc_w = in[37];    P.fc_b = in[38];
  }

  _Float16* wfrag = (_Float16*)d_ws;               // 200,704 B packed frags
  float* biasbuf = (float*)d_ws + WS_BIAS_F32_OFF; // 736 f32, contiguous after

  mtrnn_pack_weights<<<(WFRAG_F16 + 255) / 256, 256, 0, stream>>>(P, wfrag);
  mtrnn_pack_bias<<<3, 256, 0, stream>>>(P, biasbuf);
  mtrnn_persist<<<BATCH / 16, 128, LDSW_BYTES, stream>>>(data, wfrag,
                                                         (float*)d_out);
}
